// MSDeformAttn_64613488001375
// MI455X (gfx1250) — matrix operand order repaired and checked by reading, not compiled, not run on hardware
//
#include <hip/hip_runtime.h>
#include <hip/hip_bf16.h>

#define D_MODEL 128
#define N_HEADS 8
#define N_POINTS 4
#define ROWS 32           // points per block in GEMM passes
#define BTHREADS 256      // 8 waves

typedef __attribute__((ext_vector_type(16))) __bf16 v16bf;
typedef __attribute__((ext_vector_type(8)))  float  v8f;
typedef __attribute__((ext_vector_type(8)))  unsigned int v8u;   // 32-byte chunk

__device__ __forceinline__ unsigned short f2bf(float x) {
    union { float f; unsigned u; } c; c.f = x;
    unsigned u = c.u;
    unsigned r = (u + 0x7fffu + ((u >> 16) & 1u)) >> 16;   // round-to-nearest-even
    return (unsigned short)r;
}

// ---- fragment loads -------------------------------------------------------
// A fragment: 16x32 bf16 (MxK). sA row-major [ROWS][128] ushort.
// Lane layout (this code's contiguous k = 16*hi + s, not ISA table order): 16 ushorts -> 2x ds_load_b128.
__device__ __forceinline__ v16bf load_a_frag(const unsigned short* sA, int lane, int kb, int m0) {
    const int m  = m0 + (lane & 15);
    const int hi = (lane >> 4) & 1;
    union { struct { uint4 lo; uint4 hq; } q; v16bf v; } r;
    r.q.lo = *(const uint4*)&sA[m * D_MODEL + kb + hi * 16];
    r.q.hq = *(const uint4*)&sA[m * D_MODEL + kb + hi * 16 + 8];
    return r.v;
}

// B fragment: 32x16 bf16 (KxN). Weights are staged PRE-SWIZZLED in LDS as
// sWf[strip = k>>4][n][k&15] so a lane's fragment (W[kb+hi*16 .. +15][n]) is
// one contiguous 32-byte read.
__device__ __forceinline__ v16bf load_b_frag(const unsigned short* sWf, int lane, int kb, int n0) {
    const int nn    = n0 + (lane & 15);
    const int strip = (kb >> 4) + ((lane >> 4) & 1);
    union { v8u raw; v16bf v; } r;
    r.raw = *(const v8u*)&sWf[(strip * D_MODEL + nn) * 16];
    return r.v;
}

__device__ __forceinline__ void store16bf(unsigned short* dst, const unsigned short* t) {
    union { unsigned short u[16]; v8u v; } pk;
#pragma unroll
    for (int j = 0; j < 16; ++j) pk.u[j] = t[j];
    *(v8u*)dst = pk.v;
}

// stage a 128x128 k-major f32 weight matrix into swizzled bf16 LDS
__device__ __forceinline__ void stage_weight(const float* __restrict__ W, unsigned short* sWf, int tid) {
    for (int idx = tid; idx < 8 * D_MODEL; idx += BTHREADS) {
        int strip = idx >> 7, nn = idx & 127;
        unsigned short t[16];
#pragma unroll
        for (int j = 0; j < 16; ++j) t[j] = f2bf(W[(strip * 16 + j) * D_MODEL + nn]);
        store16bf(&sWf[idx * 16], t);
    }
}

// stage bf16 activation tile (ROWS x 128) from global f32, zero-padded rows
__device__ __forceinline__ void stage_act(const float* __restrict__ X, unsigned short* sA,
                                          long rowBase, int n, int tid) {
    for (int idx = tid * 4; idx < ROWS * D_MODEL; idx += BTHREADS * 4) {
        long g = rowBase + (idx >> 7);
        int  c = idx & 127;
        union { unsigned short u[4]; uint2 v; } pk;
        if (g < n) {
            float4 v = *(const float4*)&X[g * D_MODEL + c];
            pk.u[0] = f2bf(v.x); pk.u[1] = f2bf(v.y); pk.u[2] = f2bf(v.z); pk.u[3] = f2bf(v.w);
        } else {
            pk.u[0] = pk.u[1] = pk.u[2] = pk.u[3] = 0;
        }
        *(uint2*)&sA[idx] = pk.v;
    }
}

// ---------------- min/max over coords ----------------
__global__ void init_hdr_kernel(int* hdr) {
    if (threadIdx.x < 3) {
        hdr[threadIdx.x]     = 0x7FFFFFFF;      // min
        hdr[3 + threadIdx.x] = (int)0x80000000; // max
    }
}

__global__ void minmax_kernel(const int* __restrict__ coords, int* __restrict__ hdr, int n) {
    __shared__ int smn[3], smx[3];
    if (threadIdx.x < 3) { smn[threadIdx.x] = 0x7FFFFFFF; smx[threadIdx.x] = (int)0x80000000; }
    __syncthreads();
    int mn[3] = {0x7FFFFFFF, 0x7FFFFFFF, 0x7FFFFFFF};
    int mx[3] = {(int)0x80000000, (int)0x80000000, (int)0x80000000};
    for (int i = blockIdx.x * blockDim.x + threadIdx.x; i < n; i += gridDim.x * blockDim.x) {
#pragma unroll
        for (int d = 0; d < 3; ++d) {
            int c = coords[i * 3 + d];
            mn[d] = min(mn[d], c);
            mx[d] = max(mx[d], c);
        }
    }
#pragma unroll
    for (int d = 0; d < 3; ++d) { atomicMin(&smn[d], mn[d]); atomicMax(&smx[d], mx[d]); }
    __syncthreads();
    if (threadIdx.x < 3) {
        atomicMin(&hdr[threadIdx.x], smn[threadIdx.x]);
        atomicMax(&hdr[3 + threadIdx.x], smx[threadIdx.x]);
    }
}

// ---------------- Pass 1: value = value_fea @ W_val + b_val ----------------
__global__ __launch_bounds__(BTHREADS)
void value_gemm_kernel(const float* __restrict__ vfea, const float* __restrict__ Wv,
                       const float* __restrict__ bv, float* __restrict__ wsval, int n) {
    __shared__ __align__(32) unsigned short sW[D_MODEL * D_MODEL]; // 32 KB swizzled bf16
    __shared__ __align__(16) unsigned short sA[ROWS * D_MODEL];    // 8 KB bf16

    const int tid  = threadIdx.x;
    const int lane = tid & 31;
    const int wave = tid >> 5;
    const long rowBase = (long)blockIdx.x * ROWS;

    stage_weight(Wv, sW, tid);
    stage_act(vfea, sA, rowBase, n, tid);
    __syncthreads();

    const int m0 = (wave >> 2) * 16;
    const int n0 = (wave & 3) * 32;
    v8f acc0 = {}, acc1 = {};
#pragma unroll
    for (int kb = 0; kb < D_MODEL; kb += 32) {
        v16bf a  = load_a_frag(sA, lane, kb, m0);
        v16bf b0 = load_b_frag(sW, lane, kb, n0);
        v16bf b1 = load_b_frag(sW, lane, kb, n0 + 16);
        acc0 = __builtin_amdgcn_wmma_f32_16x16x32_bf16(false, a, false, b0, (short)0, acc0, false, false);
        acc1 = __builtin_amdgcn_wmma_f32_16x16x32_bf16(false, a, false, b1, (short)0, acc1, false, false);
    }
    const int hi = (lane >> 4) & 1;
    const int nc = lane & 15;
#pragma unroll
    for (int r = 0; r < 8; ++r) {
        long g = rowBase + m0 + r + hi * 8;
        if (g < n) {
            wsval[g * D_MODEL + n0 + nc]      = acc0[r] + bv[n0 + nc];
            wsval[g * D_MODEL + n0 + 16 + nc] = acc1[r] + bv[n0 + 16 + nc];
        }
    }
}

// ---------------- Pass 2: fused offsets/attn GEMM + gather + output GEMM ----------------
__global__ __launch_bounds__(BTHREADS)
void deform_attn_kernel(const float* __restrict__ query, const int* __restrict__ coords,
                        const float* __restrict__ Woff, const float* __restrict__ boff,
                        const float* __restrict__ Wattn, const float* __restrict__ battn,
                        const float* __restrict__ Wout, const float* __restrict__ bout,
                        const float* __restrict__ wsval, const int* __restrict__ hdr,
                        float* __restrict__ out, int n) {
    __shared__ __align__(32) unsigned short sW[D_MODEL * D_MODEL]; // 32 KB, reused
    __shared__ __align__(16) unsigned short sA[ROWS * D_MODEL];    // 8 KB, reused
    __shared__ float sLogit[ROWS * D_MODEL];                       // 16 KB
    __shared__ float sOut[ROWS * D_MODEL];                         // 16 KB

    const int tid  = threadIdx.x;
    const int lane = tid & 31;
    const int wave = tid >> 5;
    const long rowBase = (long)blockIdx.x * ROWS;

    int mnv[3], rc[3];
#pragma unroll
    for (int d = 0; d < 3; ++d) { mnv[d] = hdr[d]; rc[d] = hdr[3 + d] - hdr[d]; }
    const float rv0 = (float)(rc[0] / 8 + 1);
    const float rv1 = (float)(rc[1] / 8 + 1);

    // Stage 1: stage [W_off | W_attn] swizzled, query tile bf16
    for (int idx = tid; idx < 8 * D_MODEL; idx += BTHREADS) {
        int strip = idx >> 7, nn = idx & 127;
        unsigned short t[16];
        if (nn < 96) {
#pragma unroll
            for (int j = 0; j < 16; ++j) t[j] = f2bf(Woff[(strip * 16 + j) * 96 + nn]);
        } else {
#pragma unroll
            for (int j = 0; j < 16; ++j) t[j] = f2bf(Wattn[(strip * 16 + j) * 32 + (nn - 96)]);
        }
        store16bf(&sW[idx * 16], t);
    }
    stage_act(query, sA, rowBase, n, tid);
    __syncthreads();

    const int m0 = (wave >> 2) * 16;
    const int n0 = (wave & 3) * 32;
    const int hi = (lane >> 4) & 1;
    const int nc = lane & 15;

    // Stage 2: logits = query @ [W_off|W_attn] + bias -> sLogit
    {
        v8f acc0 = {}, acc1 = {};
#pragma unroll
        for (int kb = 0; kb < D_MODEL; kb += 32) {
            v16bf a  = load_a_frag(sA, lane, kb, m0);
            v16bf b0 = load_b_frag(sW, lane, kb, n0);
            v16bf b1 = load_b_frag(sW, lane, kb, n0 + 16);
            acc0 = __builtin_amdgcn_wmma_f32_16x16x32_bf16(false, a, false, b0, (short)0, acc0, false, false);
            acc1 = __builtin_amdgcn_wmma_f32_16x16x32_bf16(false, a, false, b1, (short)0, acc1, false, false);
        }
        int c0 = n0 + nc, c1 = n0 + 16 + nc;
        float bb0 = (c0 < 96) ? boff[c0] : battn[c0 - 96];
        float bb1 = (c1 < 96) ? boff[c1] : battn[c1 - 96];
#pragma unroll
        for (int r = 0; r < 8; ++r) {
            int m = m0 + r + hi * 8;
            sLogit[m * D_MODEL + c0] = acc0[r] + bb0;
            sLogit[m * D_MODEL + c1] = acc1[r] + bb1;
        }
    }
    __syncthreads();

    // Stage 3: per-(point,head) softmax + voxel quantization + gather
    {
        const int p = tid >> 3;       // 0..31 local point
        const int h = tid & 7;        // 0..7 head
        const long g = rowBase + p;
        float acc[16];
#pragma unroll
        for (int k = 0; k < 16; ++k) acc[k] = 0.0f;

        if (g < n) {
            int ref[3];
#pragma unroll
            for (int d = 0; d < 3; ++d) ref[d] = coords[g * 3 + d];

            const float* lg = &sLogit[p * D_MODEL];
            float l[N_POINTS];
#pragma unroll
            for (int pt = 0; pt < N_POINTS; ++pt) l[pt] = lg[96 + h * 4 + pt];
            float mxl = fmaxf(fmaxf(l[0], l[1]), fmaxf(l[2], l[3]));
            float e[N_POINTS], s = 0.0f;
#pragma unroll
            for (int pt = 0; pt < N_POINTS; ++pt) { e[pt] = expf(l[pt] - mxl); s += e[pt]; }
            const float inv = 1.0f / s;

#pragma unroll
            for (int pt = 0; pt < N_POINTS; ++pt) {
                float w = e[pt] * inv;
                bool ok = true;
                float idxf[3];
#pragma unroll
                for (int d = 0; d < 3; ++d) {
                    float o  = lg[h * 12 + pt * 3 + d];
                    int  oi  = (int)(o * (float)rc[d] * 0.5f);   // trunc toward 0
                    int  q   = oi >> 3;                           // floor-div by VOX
                    int  sd  = ref[d] + q * 8;
                    float ix = (float)(sd - mnv[d]) * 0.125f;
                    float ir = (float)(ref[d] - mnv[d]) * 0.125f;
                    ok = ok && (ix == ir);
                    idxf[d] = ix;
                }
                if (ok) {
                    float flat = idxf[0] * rv1 * rv0 + idxf[1] * rv0 + idxf[2];
                    int gi = (int)floorf(flat);
                    gi = min(max(gi, 0), n - 1);
                    const float4* vp = reinterpret_cast<const float4*>(wsval + (size_t)gi * D_MODEL + h * 16);
#pragma unroll
                    for (int j = 0; j < 4; ++j) {
                        float4 v = vp[j];
                        acc[4 * j + 0] += w * v.x;
                        acc[4 * j + 1] += w * v.y;
                        acc[4 * j + 2] += w * v.z;
                        acc[4 * j + 3] += w * v.w;
                    }
                }
            }
        }
        float* so = &sOut[p * D_MODEL + h * 16];
#pragma unroll
        for (int k = 0; k < 16; ++k) so[k] = acc[k];
    }
    __syncthreads();

    // Stage 4: out-tile -> bf16 sA, W_out -> swizzled sW
    for (int idx = tid * 4; idx < ROWS * D_MODEL; idx += BTHREADS * 4) {
        float4 v = *(const float4*)&sOut[idx];
        union { unsigned short u[4]; uint2 q; } pk;
        pk.u[0] = f2bf(v.x); pk.u[1] = f2bf(v.y); pk.u[2] = f2bf(v.z); pk.u[3] = f2bf(v.w);
        *(uint2*)&sA[idx] = pk.q;
    }
    stage_weight(Wout, sW, tid);
    __syncthreads();

    // Stage 5: result = out_tile @ W_out + b_out
    {
        v8f acc0 = {}, acc1 = {};
#pragma unroll
        for (int kb = 0; kb < D_MODEL; kb += 32) {
            v16bf a  = load_a_frag(sA, lane, kb, m0);
            v16bf b0 = load_b_frag(sW, lane, kb, n0);
            v16bf b1 = load_b_frag(sW, lane, kb, n0 + 16);
            acc0 = __builtin_amdgcn_wmma_f32_16x16x32_bf16(false, a, false, b0, (short)0, acc0, false, false);
            acc1 = __builtin_amdgcn_wmma_f32_16x16x32_bf16(false, a, false, b1, (short)0, acc1, false, false);
        }
#pragma unroll
        for (int r = 0; r < 8; ++r) {
            long g = rowBase + m0 + r + hi * 8;
            if (g < n) {
                out[g * D_MODEL + n0 + nc]      = acc0[r] + bout[n0 + nc];
                out[g * D_MODEL + n0 + 16 + nc] = acc1[r] + bout[n0 + 16 + nc];
            }
        }
    }
}

extern "C" void kernel_launch(void* const* d_in, const int* in_sizes, int n_in,
                              void* d_out, int out_size, void* d_ws, size_t ws_size,
                              hipStream_t stream) {
    const float* query = (const float*)d_in[0];
    const float* vfea  = (const float*)d_in[1];
    const int*   coords= (const int*)d_in[2];
    const float* Woff  = (const float*)d_in[3];
    const float* boff  = (const float*)d_in[4];
    const float* Wattn = (const float*)d_in[5];
    const float* battn = (const float*)d_in[6];
    const float* Wval  = (const float*)d_in[7];
    const float* bval  = (const float*)d_in[8];
    const float* Wout  = (const float*)d_in[9];
    const float* bout  = (const float*)d_in[10];
    float* outp = (float*)d_out;

    const int n = in_sizes[0] / D_MODEL;

    int*   hdr   = (int*)d_ws;
    float* wsval = (float*)((char*)d_ws + 256);  // n x 128 f32 value matrix

    init_hdr_kernel<<<1, 32, 0, stream>>>(hdr);
    int rblocks = (n + 255) / 256;
    if (rblocks > 1024) rblocks = 1024;
    minmax_kernel<<<rblocks, 256, 0, stream>>>(coords, hdr, n);

    const int gblocks = (n + ROWS - 1) / ROWS;
    value_gemm_kernel<<<gblocks, BTHREADS, 0, stream>>>(vfea, Wval, bval, wsval, n);
    deform_attn_kernel<<<gblocks, BTHREADS, 0, stream>>>(query, coords, Woff, boff, Wattn, battn,
                                                         Wout, bout, wsval, hdr, outp, n);
}
